// DifferentialAttention_34926674051812
// MI455X (gfx1250) — compile-verified, matrix operand order checked by reading, not run on hardware
//
#include <hip/hip_runtime.h>
#include <hip/hip_bf16.h>

typedef __attribute__((ext_vector_type(16))) __bf16 v16bf;
typedef __attribute__((ext_vector_type(8)))  float  v8f;

#define NHEAD 16
#define HDIM  64
#define BATCH 2
#define SEQ   2048
#define DMODEL 1024
#define NROWS (BATCH * SEQ)   // 4096

// ---------- helpers ----------

__device__ __forceinline__ unsigned short f2bf_raw(float f) {
    unsigned int u = __float_as_uint(f);
    unsigned int r = u + 0x7FFFu + ((u >> 16) & 1u);   // round-to-nearest-even
    return (unsigned short)(r >> 16);
}
__device__ __forceinline__ __bf16 f2bf16(float f) {
    unsigned short s = f2bf_raw(f);
    return __builtin_bit_cast(__bf16, s);
}

// 16-bit A-matrix 16x32 layout (ISA 7.12.2): lane L holds row M=L&15,
// lane-half g=L>>4. VGPR i<4 holds K = {2i, 2i+1} + 8g ; VGPR i>=4 holds
// K = {2i+8, 2i+9} + 8g.  B (32x16) mirrors with lane = column.
__device__ __forceinline__ int frag_k(int i, int g) {
    return ((i < 4) ? (2 * i) : (2 * i + 8)) + 8 * g;
}

union AF  { v16bf v; __bf16 e[16]; unsigned int u[8]; };
union CF  { v8f   v; float  e[8]; };

#define WMMA_BF16(A, B, C) \
    __builtin_amdgcn_wmma_f32_16x16x32_bf16(false, (A), false, (B), (short)0, (C), false, false)

// CDNA5 async global->LDS 128-bit copy (ASYNCcnt-tracked, no VGPR staging).
__device__ __forceinline__ void async_g2l_b128(void* lds_dst, const void* src) {
    unsigned lds_off = (unsigned)(uintptr_t)lds_dst;            // LDS byte offset
    unsigned long long ga = (unsigned long long)(uintptr_t)src; // 64-bit global addr
    asm volatile("global_load_async_to_lds_b128 %0, %1, off"
                 :: "v"(lds_off), "v"(ga) : "memory");
}
__device__ __forceinline__ void wait_async0() {
    asm volatile("s_wait_asynccnt 0" ::: "memory");
}
__device__ __forceinline__ void wait_ds0() {
    asm volatile("s_wait_dscnt 0" ::: "memory");
}

// ---------- lambda scalar ----------

__global__ void lam_kernel(const float* __restrict__ lq1, const float* __restrict__ lk1,
                           const float* __restrict__ lq2, const float* __restrict__ lk2,
                           float* __restrict__ out) {
    __shared__ float s1[64], s2[64];
    int t = threadIdx.x;
    s1[t] = lq1[t] * lk1[t];
    s2[t] = lq2[t] * lk2[t];
    __syncthreads();
    for (int off = 32; off > 0; off >>= 1) {
        if (t < off) { s1[t] += s1[t + off]; s2[t] += s2[t + off]; }
        __syncthreads();
    }
    if (t == 0) out[0] = __expf(s1[0]) - __expf(s2[0]) + 0.8f;
}

// ---------- fp32 -> bf16 (plain) ----------

__global__ void cvt_f32_bf16(const float* __restrict__ in, __bf16* __restrict__ out, int n) {
    int i = blockIdx.x * blockDim.x + threadIdx.x;
    if (i < n) out[i] = f2bf16(in[i]);
}

// ---------- fp32 [rows][cols] -> bf16 transposed [cols][rows] ----------
// grid: (cols/256, rows), 256 threads. Coalesced reads.

__global__ void cvt_f32_bf16_t(const float* __restrict__ in, __bf16* __restrict__ out,
                               int rows, int cols) {
    int c = blockIdx.x * 256 + threadIdx.x;
    int r = blockIdx.y;
    out[(size_t)c * rows + r] = f2bf16(in[(size_t)r * cols + c]);
}

// ---------- bf16 WMMA GEMM: out[M,Nc] = A[M,K] @ Wt[Nc,K]^T + bias ----------
// Wt is stored transposed so both A- and B-fragments read contiguous K-pairs.
// 128 threads = 4 waves; 64x64 output tile; K step 64; async global->LDS.
// OUT_MODE: 0 = f32 [M,Nc], 1 = bf16 [M,Nc], 2 = bf16 transposed [Nc,M].

template <int OUT_MODE>
__global__ __launch_bounds__(128) void gemm_bf16(const __bf16* __restrict__ A,
                                                 const __bf16* __restrict__ Wt,
                                                 const float* __restrict__ bias,
                                                 void* __restrict__ outp,
                                                 int M, int K, int Nc) {
    const int lane = threadIdx.x & 31;
    const int wave = threadIdx.x >> 5;
    const int g    = lane >> 4;
    const int ln   = lane & 15;
    const int m0   = blockIdx.x * 64;
    const int n0   = blockIdx.y * 64;

    __shared__ __bf16 As[64][72];   // row stride 144B (16B multiple)
    __shared__ __bf16 Ws[64][72];   // [n][k] layout

    v8f acc[4];
#pragma unroll
    for (int c = 0; c < 4; ++c) acc[c] = (v8f){0,0,0,0,0,0,0,0};

    for (int k0 = 0; k0 < K; k0 += 64) {
        // A tile 64x64 and Wt tile 64x64 : 512 16B chunks each, async to LDS
        for (int q = threadIdx.x; q < 512; q += 128) {
            int r = q >> 3, c = (q & 7) * 8;
            async_g2l_b128(&As[r][c], A  + (size_t)(m0 + r) * K + k0 + c);
            async_g2l_b128(&Ws[r][c], Wt + (size_t)(n0 + r) * K + k0 + c);
        }
        wait_async0();
        __syncthreads();

#pragma unroll
        for (int kh = 0; kh < 2; ++kh) {
            AF a;
#pragma unroll
            for (int i = 0; i < 8; ++i)
                a.u[i] = *(const unsigned int*)&As[wave * 16 + ln][kh * 32 + frag_k(i, g)];
#pragma unroll
            for (int c = 0; c < 4; ++c) {
                AF b;
#pragma unroll
                for (int i = 0; i < 8; ++i)
                    b.u[i] = *(const unsigned int*)&Ws[c * 16 + ln][kh * 32 + frag_k(i, g)];
                acc[c] = WMMA_BF16(a.v, b.v, acc[c]);
            }
        }
        __syncthreads();
    }

#pragma unroll
    for (int c = 0; c < 4; ++c) {
        CF o; o.v = acc[c];
#pragma unroll
        for (int r = 0; r < 8; ++r) {
            int row = m0 + wave * 16 + r + 8 * g;
            int col = n0 + c * 16 + ln;
            float val = o.e[r] + bias[col];
            if (OUT_MODE == 0)
                ((float*)outp)[(size_t)row * Nc + col] = val;
            else if (OUT_MODE == 1)
                ((__bf16*)outp)[(size_t)row * Nc + col] = f2bf16(val);
            else
                ((__bf16*)outp)[(size_t)col * M + row] = f2bf16(val);   // transposed
        }
    }
}

// ---------- flash-style differential attention ----------
// grid (S/64, NH, B), 128 threads. Q/K: [N, 2D] bf16 (q1|q2 split at D),
// Vt: [D, N] bf16 (transposed), output O: [N, D] bf16 (head-concatenated).

__global__ __launch_bounds__(128) void diff_attn(const __bf16* __restrict__ Q,
                                                 const __bf16* __restrict__ Kmat,
                                                 const __bf16* __restrict__ Vt,
                                                 const float* __restrict__ mask,
                                                 const float* __restrict__ lam_p,
                                                 __bf16* __restrict__ O) {
    const int lane = threadIdx.x & 31;
    const int wave = threadIdx.x >> 5;
    const int g    = lane >> 4;
    const int ln   = lane & 15;
    const int q0   = blockIdx.x * 64;
    const int h    = blockIdx.y;
    const int b    = blockIdx.z;
    const float scale = 0.125f;   // 64^-0.5

    __shared__ __bf16 k1s[32][72];    // [kv][d]
    __shared__ __bf16 k2s[32][72];
    __shared__ __bf16 vts[64][40];    // [d][kv]  (from transposed V)
    __shared__ __bf16 qs[64][72];
    __shared__ __bf16 ps[4][16][40];  // per-wave prob tile (C->A relayout)
    __shared__ float  maskadd[32];

    // ---- stage q1 (64x64) then q2; extract per-wave A-fragments (K=64 -> 2) ----
    AF q1f[2], q2f[2];
    for (int t = threadIdx.x; t < 512; t += 128) {
        int r = t >> 3, c = (t & 7) * 8;
        async_g2l_b128(&qs[r][c],
            Q + (size_t)(b * SEQ + q0 + r) * (2 * DMODEL) + h * HDIM + c);
    }
    wait_async0();
    __syncthreads();
#pragma unroll
    for (int kh = 0; kh < 2; ++kh)
#pragma unroll
        for (int i = 0; i < 8; ++i)
            q1f[kh].u[i] = *(const unsigned int*)&qs[wave * 16 + ln][kh * 32 + frag_k(i, g)];
    __syncthreads();
    for (int t = threadIdx.x; t < 512; t += 128) {
        int r = t >> 3, c = (t & 7) * 8;
        async_g2l_b128(&qs[r][c],
            Q + (size_t)(b * SEQ + q0 + r) * (2 * DMODEL) + DMODEL + h * HDIM + c);
    }
    wait_async0();
    __syncthreads();
#pragma unroll
    for (int kh = 0; kh < 2; ++kh)
#pragma unroll
        for (int i = 0; i < 8; ++i)
            q2f[kh].u[i] = *(const unsigned int*)&qs[wave * 16 + ln][kh * 32 + frag_k(i, g)];
    __syncthreads();

    // ---- online-softmax state (rows r+8g of this wave's 16-row tile) ----
    float m1[8], l1[8], m2[8], l2[8];
    v8f o1[4], o2[4];
#pragma unroll
    for (int r = 0; r < 8; ++r) { m1[r] = -1.0e30f; l1[r] = 0.0f; m2[r] = -1.0e30f; l2[r] = 0.0f; }
#pragma unroll
    for (int c = 0; c < 4; ++c) { o1[c] = (v8f){0,0,0,0,0,0,0,0}; o2[c] = (v8f){0,0,0,0,0,0,0,0}; }

    for (int kv0 = 0; kv0 < SEQ; kv0 += 32) {
        // async tile loads: k1/k2 [32x64], vt [64x32]
        for (int t = threadIdx.x; t < 256; t += 128) {
            int r = t >> 3, c = (t & 7) * 8;
            size_t rowq = (size_t)(b * SEQ + kv0 + r);
            async_g2l_b128(&k1s[r][c], Kmat + rowq * (2 * DMODEL) + h * HDIM + c);
            async_g2l_b128(&k2s[r][c], Kmat + rowq * (2 * DMODEL) + DMODEL + h * HDIM + c);
        }
        for (int t = threadIdx.x; t < 256; t += 128) {
            int r = t >> 2, c = (t & 3) * 8;
            async_g2l_b128(&vts[r][c],
                Vt + (size_t)(h * HDIM + r) * NROWS + b * SEQ + kv0 + c);
        }
        if (threadIdx.x < 32)
            maskadd[threadIdx.x] = (1.0f - mask[b * SEQ + kv0 + threadIdx.x]) * -10000.0f;
        wait_async0();
        __syncthreads();

        // ---- scores: two 16x16 tiles per stream, K=64 (2 WMMA each) ----
        v8f s1t[2], s2t[2];
#pragma unroll
        for (int tn = 0; tn < 2; ++tn) { s1t[tn] = (v8f){0,0,0,0,0,0,0,0}; s2t[tn] = (v8f){0,0,0,0,0,0,0,0}; }
#pragma unroll
        for (int tn = 0; tn < 2; ++tn) {
#pragma unroll
            for (int kh = 0; kh < 2; ++kh) {
                AF b1, b2;   // B[k=d, n=col] = K[kv0+col, d] (contiguous along k1s row)
#pragma unroll
                for (int i = 0; i < 8; ++i) {
                    b1.u[i] = *(const unsigned int*)&k1s[tn * 16 + ln][kh * 32 + frag_k(i, g)];
                    b2.u[i] = *(const unsigned int*)&k2s[tn * 16 + ln][kh * 32 + frag_k(i, g)];
                }
                s1t[tn] = WMMA_BF16(q1f[kh].v, b1.v, s1t[tn]);
                s2t[tn] = WMMA_BF16(q2f[kh].v, b2.v, s2t[tn]);
            }
        }

        // ---- two independent online softmaxes + PV ----
#pragma unroll
        for (int stream_i = 0; stream_i < 2; ++stream_i) {
            CF t0, t1;
            t0.v = stream_i ? s2t[0] : s1t[0];
            t1.v = stream_i ? s2t[1] : s1t[1];
            float* mm = stream_i ? m2 : m1;
            float* ll = stream_i ? l2 : l1;
            v8f*   oo = stream_i ? o2 : o1;

            float p0[8], p1[8], alpha[8];
#pragma unroll
            for (int r = 0; r < 8; ++r) {
                float x0 = t0.e[r] * scale + maskadd[ln];
                float x1 = t1.e[r] * scale + maskadd[16 + ln];
                float mx = fmaxf(x0, x1);
#pragma unroll
                for (int off = 1; off < 16; off <<= 1)
                    mx = fmaxf(mx, __shfl_xor(mx, off, 32));
                float mnew = fmaxf(mm[r], mx);
                float a  = __expf(mm[r] - mnew);
                float e0 = __expf(x0 - mnew);
                float e1 = __expf(x1 - mnew);
                float rs = e0 + e1;
#pragma unroll
                for (int off = 1; off < 16; off <<= 1)
                    rs += __shfl_xor(rs, off, 32);
                ll[r] = ll[r] * a + rs;
                mm[r] = mnew;
                alpha[r] = a; p0[r] = e0; p1[r] = e1;
            }
#pragma unroll
            for (int c = 0; c < 4; ++c) {
                CF z; z.v = oo[c];
#pragma unroll
                for (int r = 0; r < 8; ++r) z.e[r] *= alpha[r];
                oo[c] = z.v;
            }
            // C-layout -> A-layout via per-wave LDS round trip
#pragma unroll
            for (int r = 0; r < 8; ++r) {
                int row = r + 8 * g;
                ps[wave][row][ln]      = f2bf16(p0[r]);
                ps[wave][row][16 + ln] = f2bf16(p1[r]);
            }
            wait_ds0();
            AF pf;
#pragma unroll
            for (int i = 0; i < 8; ++i)
                pf.u[i] = *(const unsigned int*)&ps[wave][ln][frag_k(i, g)];
            // PV: out(16x64) += P(16x32) @ V(32x64); V-frags contiguous in vts rows
#pragma unroll
            for (int c = 0; c < 4; ++c) {
                AF vb;
#pragma unroll
                for (int i = 0; i < 8; ++i)
                    vb.u[i] = *(const unsigned int*)&vts[c * 16 + ln][frag_k(i, g)];
                oo[c] = WMMA_BF16(pf.v, vb.v, oo[c]);
            }
        }
        __syncthreads();
    }

    // ---- epilogue: o1/l1 - lam * o2/l2, head-concatenated bf16 ----
    const float lam = *lam_p;
#pragma unroll
    for (int c = 0; c < 4; ++c) {
        CF a1, a2; a1.v = o1[c]; a2.v = o2[c];
#pragma unroll
        for (int r = 0; r < 8; ++r) {
            int row = q0 + wave * 16 + r + 8 * g;
            float val = a1.e[r] / l1[r] - lam * (a2.e[r] / l2[r]);
            O[(size_t)(b * SEQ + row) * DMODEL + h * HDIM + c * 16 + ln] = f2bf16(val);
        }
    }
}

// ---------- launch ----------

extern "C" void kernel_launch(void* const* d_in, const int* in_sizes, int n_in,
                              void* d_out, int out_size, void* d_ws, size_t ws_size,
                              hipStream_t stream) {
    const float* X    = (const float*)d_in[0];
    const float* mask = (const float*)d_in[1];
    const float* Wq   = (const float*)d_in[2];
    const float* bq   = (const float*)d_in[3];
    const float* Wk   = (const float*)d_in[4];
    const float* bk   = (const float*)d_in[5];
    const float* Wv   = (const float*)d_in[6];
    const float* bv   = (const float*)d_in[7];
    const float* Wo   = (const float*)d_in[8];
    const float* bo   = (const float*)d_in[9];
    const float* lq1  = (const float*)d_in[10];
    const float* lk1  = (const float*)d_in[11];
    const float* lq2  = (const float*)d_in[12];
    const float* lk2  = (const float*)d_in[13];

    char* ws = (char*)d_ws;
    size_t off = 0;
    auto alloc = [&](size_t bytes) -> void* {
        void* p = ws + off;
        off += (bytes + 255) & ~(size_t)255;
        return p;
    };
    float*  lam  = (float*) alloc(sizeof(float));
    __bf16* Xb   = (__bf16*)alloc((size_t)NROWS * DMODEL * 2);
    __bf16* Wqt  = (__bf16*)alloc((size_t)DMODEL * 2 * DMODEL * 2);   // [2D][D]
    __bf16* Wkt  = (__bf16*)alloc((size_t)DMODEL * 2 * DMODEL * 2);
    __bf16* Wvt  = (__bf16*)alloc((size_t)DMODEL * DMODEL * 2);       // [D][D]
    __bf16* Wot  = (__bf16*)alloc((size_t)DMODEL * DMODEL * 2);
    __bf16* Qb   = (__bf16*)alloc((size_t)NROWS * 2 * DMODEL * 2);
    __bf16* Kb   = (__bf16*)alloc((size_t)NROWS * 2 * DMODEL * 2);
    __bf16* Vtb  = (__bf16*)alloc((size_t)NROWS * DMODEL * 2);        // [D][N]
    __bf16* Ab   = (__bf16*)alloc((size_t)NROWS * DMODEL * 2);

    lam_kernel<<<1, 64, 0, stream>>>(lq1, lk1, lq2, lk2, lam);

    cvt_f32_bf16<<<(NROWS * DMODEL + 255) / 256, 256, 0, stream>>>(X, Xb, NROWS * DMODEL);
    // weights -> bf16, transposed to [Nc][K]
    cvt_f32_bf16_t<<<dim3((2 * DMODEL) / 256, DMODEL), 256, 0, stream>>>(Wq, Wqt, DMODEL, 2 * DMODEL);
    cvt_f32_bf16_t<<<dim3((2 * DMODEL) / 256, DMODEL), 256, 0, stream>>>(Wk, Wkt, DMODEL, 2 * DMODEL);
    cvt_f32_bf16_t<<<dim3(DMODEL / 256, DMODEL), 256, 0, stream>>>(Wv, Wvt, DMODEL, DMODEL);
    cvt_f32_bf16_t<<<dim3(DMODEL / 256, DMODEL), 256, 0, stream>>>(Wo, Wot, DMODEL, DMODEL);

    gemm_bf16<1><<<dim3(NROWS / 64, (2 * DMODEL) / 64), 128, 0, stream>>>(
        Xb, Wqt, bq, Qb, NROWS, DMODEL, 2 * DMODEL);
    gemm_bf16<1><<<dim3(NROWS / 64, (2 * DMODEL) / 64), 128, 0, stream>>>(
        Xb, Wkt, bk, Kb, NROWS, DMODEL, 2 * DMODEL);
    gemm_bf16<2><<<dim3(NROWS / 64, DMODEL / 64), 128, 0, stream>>>(
        Xb, Wvt, bv, Vtb, NROWS, DMODEL, DMODEL);   // V stored transposed [D][N]

    diff_attn<<<dim3(SEQ / 64, NHEAD, BATCH), 128, 0, stream>>>(Qb, Kb, Vtb, mask, lam, Ab);

    gemm_bf16<0><<<dim3(NROWS / 64, DMODEL / 64), 128, 0, stream>>>(
        Ab, Wot, bo, d_out, NROWS, DMODEL, DMODEL);
}
